// LSTMFrameSRVSemantic_53360673686034
// MI455X (gfx1250) — matrix operand order repaired and checked by reading, not compiled, not run on hardware
//
#include <hip/hip_runtime.h>
#include <hip/hip_bf16.h>
#include <math.h>

// ---------------------------------------------------------------- types
typedef __attribute__((ext_vector_type(16))) __bf16 v16bf;
typedef __attribute__((ext_vector_type(8)))  float  v8f;

struct __attribute__((aligned(16))) U4 { unsigned int x, y, z, w; };
union V16 { U4 q[2]; v16bf v; unsigned short s[16]; };

__device__ __forceinline__ float bf2f(unsigned short u) {
    unsigned int x = ((unsigned int)u) << 16;
    float f; __builtin_memcpy(&f, &x, 4); return f;
}
__device__ __forceinline__ unsigned short f2bf(float f) {
    unsigned int x; __builtin_memcpy(&x, &f, 4);
    unsigned int lsb = (x >> 16) & 1u;
    x += 0x7fffu + lsb;
    return (unsigned short)(x >> 16);
}

// hardware transcendentals (gfx1250 has single-instruction exp2/rcp/rsq/tanh)
#define LOG2E 1.44269504088896340736f
__device__ __forceinline__ float fast_sigmoid(float x) {
    float e = __builtin_amdgcn_exp2f(-x * LOG2E);        // v_exp_f32
    return __builtin_amdgcn_rcpf(1.0f + e);              // v_rcp_f32 (no div fixup)
}
__device__ __forceinline__ float fast_tanh(float x) {
#if __has_builtin(__builtin_amdgcn_tanhf)
    return __builtin_amdgcn_tanhf(x);                    // v_tanh_f32
#elif __has_builtin(__builtin_amdgcn_tanh_f32)
    return __builtin_amdgcn_tanh_f32(x);
#else
    // tanh(x) = 1 - 2/(e^{2x}+1), via hardware exp2 + rcp
    float e = __builtin_amdgcn_exp2f(2.0f * x * LOG2E);
    return 1.0f - 2.0f * __builtin_amdgcn_rcpf(e + 1.0f);
#endif
}

// ---------------------------------------------------------------- weight swizzle
// dst tile layout: [ntile][ktile][lane(32)][slot(16)] bf16
// B element (k,n): lane = (n%16) + 16*((k%16)/8 within tile), slot = k%8 + 8*(k/16 within tile)
__global__ void k_swizzleB(unsigned short* dst, const float* W, int Nn, int KtTotal,
                           int ktOff, int KtThis, int Ksrc, int ldW, int total) {
    for (int idx = blockIdx.x * blockDim.x + threadIdx.x; idx < total;
         idx += gridDim.x * blockDim.x) {
        int slot = idx & 15;
        int lane = (idx >> 4) & 31;
        int rest = idx >> 9;
        int ktl  = rest % KtThis;
        int nt   = rest / KtThis;
        int k = 32 * ktl + 8 * (lane >> 4) + (slot & 7) + 16 * (slot >> 3);
        int n = 16 * nt + (lane & 15);
        float v = (k < Ksrc) ? W[(size_t)n * ldW + k] : 0.0f;
        dst[(((size_t)(nt * KtTotal + ktOff + ktl)) * 32 + lane) * 16 + slot] = f2bf(v);
    }
}

__global__ void k_init_sync(int* p) { p[0] = 0; p[1] = 0; }

// ---------------------------------------------------------------- grid barrier
__device__ __forceinline__ void grid_sync(int* cnt, int* gen, int nb) {
    __syncthreads();
    if (threadIdx.x == 0) {
        int g = __hip_atomic_load(gen, __ATOMIC_RELAXED, __HIP_MEMORY_SCOPE_AGENT);
        int a = __hip_atomic_fetch_add(cnt, 1, __ATOMIC_ACQ_REL, __HIP_MEMORY_SCOPE_AGENT);
        if (a == nb - 1) {
            __hip_atomic_store(cnt, 0, __ATOMIC_RELAXED, __HIP_MEMORY_SCOPE_AGENT);
            __hip_atomic_fetch_add(gen, 1, __ATOMIC_ACQ_REL, __HIP_MEMORY_SCOPE_AGENT);
        } else {
            while (__hip_atomic_load(gen, __ATOMIC_ACQUIRE, __HIP_MEMORY_SCOPE_AGENT) == g)
                __builtin_amdgcn_s_sleep(1);
        }
    }
    __syncthreads();
}

// ---------------------------------------------------------------- frame LSTM
// 32 blocks x 128 threads (4 waves). Block b owns hidden cols [16b,16b+16),
// wave g computes gate g for those 16 columns. K = 1024 = [x_t | h_{t-1}].
// Weights LDS-resident; per-step grid sync to broadcast h.
__global__ void __launch_bounds__(128)
k_frame_lstm(const float* __restrict__ x,
             const unsigned short* fWc0, const unsigned short* fWc1,
             const unsigned short* fWc2,
             const float* fb0, const float* fb1, const float* fb2,
             unsigned short* hb0, unsigned short* hb1,
             int* syncCnt, int* syncGen, int T) {
    extern __shared__ char smem[];
    unsigned short* Wlds = (unsigned short*)smem;            // 4*32*512 = 65536 el
    unsigned short* Alds = Wlds + 65536;                     // 32*512   = 16384 el
    float* Gex  = (float*)(smem + 163840);                   // 4*256
    float* Cst  = Gex + 1024;                                // 256
    float* bLds = Cst + 256;                                 // 64

    const int tid  = threadIdx.x;
    const int b    = blockIdx.x;           // 0..31
    const int wave = tid >> 5;             // gate id 0..3
    const int lane = tid & 31;
    const int half = lane >> 4;
    const int nloc = lane & 15;

    for (int layer = 0; layer < 3; ++layer) {
        const unsigned short* Wsw = (layer == 0) ? fWc0 : (layer == 1) ? fWc1 : fWc2;
        const float* Bias = (layer == 0) ? fb0 : (layer == 1) ? fb1 : fb2;
        const unsigned short* hin = (layer == 1) ? hb0 : hb1;   // layer0 unused
        unsigned short* hout      = (layer == 1) ? hb1 : hb0;

        // load this block's weight slice into LDS (4 gate ntiles, 16384 ushorts each)
        for (int g = 0; g < 4; ++g) {
            const U4* s4 = (const U4*)(Wsw + (size_t)(32 * g + b) * 16384);
            U4* d4 = (U4*)(Wlds + g * 16384);
            for (int i = tid; i < 2048; i += 128) d4[i] = s4[i];
        }
        for (int i = tid; i < 64; i += 128)
            bLds[i] = Bias[512 * (i >> 4) + 16 * b + (i & 15)];
        for (int i = tid; i < 256; i += 128) Cst[i] = 0.0f;
        __syncthreads();

        for (int t = 0; t < T; ++t) {
            // ---- stage A = [x_t | h_{t-1}] in WMMA-A swizzle: [kt][lane][slot]
            for (int p = tid; p < 1024; p += 128) {
                int kt = p >> 5, ln = p & 31;
                int hf = ln >> 4, m = ln & 15;
                int k0a = 32 * kt + 8 * hf;
                U4* dst = (U4*)(Alds + p * 16);
#pragma unroll
                for (int run = 0; run < 2; ++run) {
                    int k0 = k0a + 16 * run;
                    U4 val;
                    if (k0 >= 512) {          // recurrent half
                        int j0 = k0 - 512;
                        if (t == 0) { val.x = val.y = val.z = val.w = 0u; }
                        else val = *(const U4*)(hout + ((size_t)m * T + (t - 1)) * 512 + j0);
                    } else if (layer == 0) {  // raw input, D=80 padded to 512
                        unsigned short tmp[8];
#pragma unroll
                        for (int e = 0; e < 8; ++e) {
                            int kk = k0 + e;
                            float xv = (kk < 80) ? x[((size_t)m * T + t) * 80 + kk] : 0.0f;
                            tmp[e] = f2bf(xv);
                        }
                        __builtin_memcpy(&val, tmp, 16);
                    } else {                  // previous layer h
                        val = *(const U4*)(hin + ((size_t)m * T + t) * 512 + k0);
                    }
                    dst[run] = val;
                }
            }
            __syncthreads();

            // ---- WMMA: gate `wave`, 16x16 out, K=1024
            v8f acc = {};
#pragma unroll 4
            for (int kt = 0; kt < 32; ++kt) {
                V16 a, bm;
                const U4* ap = (const U4*)(Alds + kt * 512 + lane * 16);
                a.q[0] = ap[0]; a.q[1] = ap[1];
                const U4* bp = (const U4*)(Wlds + (wave * 32 + kt) * 512 + lane * 16);
                bm.q[0] = bp[0]; bm.q[1] = bp[1];
                acc = __builtin_amdgcn_wmma_f32_16x16x32_bf16(
                        false, a.v, false, bm.v, (short)0, acc, false, false);
            }
#pragma unroll
            for (int r = 0; r < 8; ++r)
                Gex[wave * 256 + (r + 8 * half) * 16 + nloc] = acc[r];
            __syncthreads();

            // ---- gate nonlinearities + state update (block's 16 hidden cols)
            for (int e = tid; e < 256; e += 128) {
                int m = e >> 4, n = e & 15;
                float gi = Gex[e]       + bLds[n];
                float gf = Gex[256 + e] + bLds[16 + n];
                float gg = Gex[512 + e] + bLds[32 + n];
                float go = Gex[768 + e] + bLds[48 + n];
                float c = fast_sigmoid(gf) * Cst[e] + fast_sigmoid(gi) * fast_tanh(gg);
                Cst[e] = c;
                float h = fast_sigmoid(go) * fast_tanh(c);
                hout[((size_t)m * T + t) * 512 + 16 * b + n] = f2bf(h);
            }
            __threadfence();
            grid_sync(syncCnt, syncGen, 32);
        }
    }
}

// ---------------------------------------------------------------- generic WMMA GEMM
// C[M,N] = rowmask(A[M,K] bf16) @ Bsw + bias.   Masked rows -> bias only.
__global__ void __launch_bounds__(256)
k_gemm(const unsigned short* __restrict__ A, const unsigned short* __restrict__ Bsw,
       const float* __restrict__ bias, float* __restrict__ Cout,
       int M, int N, int K, const int* lens, int rowsPerBatch, int lensMin1) {
    int Nt = N >> 4, Kt = K >> 5, Mt = M >> 4;
    int wt = blockIdx.x * (blockDim.x >> 5) + (threadIdx.x >> 5);
    if (wt >= Mt * Nt) return;                       // wave-uniform
    int mtile = wt / Nt, ntile = wt % Nt;
    int lane = threadIdx.x & 31, half = lane >> 4, nl = lane & 15;
    int row = 16 * mtile + nl;
    bool valid = true;
    if (lens) {
        int L = lens[row / rowsPerBatch];
        if (lensMin1 && L < 1) L = 1;
        valid = (row % rowsPerBatch) < L;
    }
    const unsigned short* Arow = A + (size_t)row * K;
    v8f acc = {};
    for (int kt = 0; kt < Kt; ++kt) {
        V16 a, bm;
        if (valid) {
            int k0 = 32 * kt + 8 * half;
            a.q[0] = *(const U4*)(Arow + k0);
            a.q[1] = *(const U4*)(Arow + k0 + 16);
        } else {
            a.q[0].x = a.q[0].y = a.q[0].z = a.q[0].w = 0u;
            a.q[1] = a.q[0];
        }
        const U4* bp = (const U4*)(Bsw + (((size_t)(ntile * Kt + kt)) * 32 + lane) * 16);
        bm.q[0] = bp[0]; bm.q[1] = bp[1];
        acc = __builtin_amdgcn_wmma_f32_16x16x32_bf16(
                false, a.v, false, bm.v, (short)0, acc, false, false);
    }
    int n = 16 * ntile + nl;
    float bv = bias ? bias[n] : 0.0f;
#pragma unroll
    for (int r = 0; r < 8; ++r) {
        int m = 16 * mtile + r + 8 * half;
        Cout[(size_t)m * N + n] = acc[r] + bv;
    }
}

// ---------------------------------------------------------------- ragged tail means
__global__ void k_reps(const unsigned short* __restrict__ h, const int* x_lens,
                       const int* wstart, const int* wend, const int* wlens,
                       const int* tailp, unsigned short* reps, int T, int W) {
    int bw = blockIdx.x;
    int b = bw / W, w = bw % W;
    int we = wend[bw];
    int xl = x_lens[b];
    int ef = we < xl ? we : xl;
    int sf = wstart[bw] > 0 ? wstart[bw] : 0;
    int i0 = ef - tailp[0] > sf ? ef - tailp[0] : sf;
    bool valid = (w < wlens[b]) && (we > 0) && (ef > sf) && (ef > i0);
    for (int col = threadIdx.x; col < 512; col += blockDim.x) {
        float s = 0.0f;
        if (valid) {
            for (int t = i0; t < ef; ++t)
                s += bf2f(h[((size_t)b * T + t) * 512 + col]);
            s /= (float)(ef - i0);
        }
        reps[(size_t)bw * 512 + col] = f2bf(s);
    }
}

// ---------------------------------------------------------------- LayerNorm + GELU
__global__ void __launch_bounds__(256)
k_ln_gelu(const float* __restrict__ r, const float* lnW, const float* lnB,
          unsigned short* out) {
    __shared__ float red[256];
    int row = blockIdx.x, tid = threadIdx.x;
    float v = r[(size_t)row * 256 + tid];
    red[tid] = v; __syncthreads();
    for (int s = 128; s > 0; s >>= 1) { if (tid < s) red[tid] += red[tid + s]; __syncthreads(); }
    float mu = red[0] / 256.0f; __syncthreads();
    float d = v - mu;
    red[tid] = d * d; __syncthreads();
    for (int s = 128; s > 0; s >>= 1) { if (tid < s) red[tid] += red[tid + s]; __syncthreads(); }
    float var = red[0] / 256.0f;
    float y = d * __builtin_amdgcn_rsqf(var + 1e-5f) * lnW[tid] + lnB[tid];
    float g = 0.5f * y * (1.0f + erff(y * 0.70710678118654752f));  // exact GELU
    out[(size_t)row * 256 + tid] = f2bf(g);
}

// ---------------------------------------------------------------- word LSTM
// Single block, 16 waves. Wave v owns hidden cols [16v,16v+16) and computes all
// 4 gate fragments itself -> c/h update entirely in accumulator registers.
__global__ void __launch_bounds__(512)
k_word_lstm(const unsigned short* __restrict__ rin,
            const unsigned short* __restrict__ wWsw,
            const float* __restrict__ wb, unsigned short* hw, int W) {
    __shared__ unsigned short Alds[8192];    // [kt(16)][lane(32)][slot(16)]
    __shared__ unsigned short hstage[4096];  // [m(16)][j(256)]
    __shared__ float biasW[1024];
    int tid = threadIdx.x;
    int wave = tid >> 5, lane = tid & 31, half = lane >> 4, nl = lane & 15;
    for (int i = tid; i < 1024; i += 512) biasW[i] = wb[i];
    for (int i = tid; i < 4096; i += 512) hstage[i] = 0;
    __syncthreads();
    float cst[8];
#pragma unroll
    for (int r = 0; r < 8; ++r) cst[r] = 0.0f;
    int j = 16 * wave + nl;

    for (int t = 0; t < W; ++t) {
        { // stage A = [r_t | h_{t-1}], one (kt,lane) pair per thread
            int kt = tid >> 5, ln = tid & 31;
            int hf = ln >> 4, m = ln & 15;
            U4* dst = (U4*)(Alds + (kt * 32 + ln) * 16);
#pragma unroll
            for (int run = 0; run < 2; ++run) {
                int k0 = 32 * kt + 8 * hf + 16 * run;
                U4 val;
                if (k0 < 256)
                    val = *(const U4*)(rin + ((size_t)m * W + t) * 256 + k0);
                else
                    val = *(const U4*)(hstage + m * 256 + (k0 - 256));
                dst[run] = val;
            }
        }
        __syncthreads();

        v8f a0 = {}, a1 = {}, a2 = {}, a3 = {};
        for (int kt = 0; kt < 16; ++kt) {
            V16 a;
            const U4* ap = (const U4*)(Alds + (kt * 32 + lane) * 16);
            a.q[0] = ap[0]; a.q[1] = ap[1];
            V16 b0, b1, b2, b3;
            const U4* p0 = (const U4*)(wWsw + (((size_t)((0 * 16 + wave) * 16 + kt)) * 32 + lane) * 16);
            const U4* p1 = (const U4*)(wWsw + (((size_t)((1 * 16 + wave) * 16 + kt)) * 32 + lane) * 16);
            const U4* p2 = (const U4*)(wWsw + (((size_t)((2 * 16 + wave) * 16 + kt)) * 32 + lane) * 16);
            const U4* p3 = (const U4*)(wWsw + (((size_t)((3 * 16 + wave) * 16 + kt)) * 32 + lane) * 16);
            b0.q[0] = p0[0]; b0.q[1] = p0[1];
            b1.q[0] = p1[0]; b1.q[1] = p1[1];
            b2.q[0] = p2[0]; b2.q[1] = p2[1];
            b3.q[0] = p3[0]; b3.q[1] = p3[1];
            a0 = __builtin_amdgcn_wmma_f32_16x16x32_bf16(false, a.v, false, b0.v, (short)0, a0, false, false);
            a1 = __builtin_amdgcn_wmma_f32_16x16x32_bf16(false, a.v, false, b1.v, (short)0, a1, false, false);
            a2 = __builtin_amdgcn_wmma_f32_16x16x32_bf16(false, a.v, false, b2.v, (short)0, a2, false, false);
            a3 = __builtin_amdgcn_wmma_f32_16x16x32_bf16(false, a.v, false, b3.v, (short)0, a3, false, false);
        }
#pragma unroll
        for (int r = 0; r < 8; ++r) {
            int m = r + 8 * half;
            float gi = a0[r] + biasW[j];
            float gf = a1[r] + biasW[256 + j];
            float gg = a2[r] + biasW[512 + j];
            float go = a3[r] + biasW[768 + j];
            float c = fast_sigmoid(gf) * cst[r] + fast_sigmoid(gi) * fast_tanh(gg);
            cst[r] = c;
            float h = fast_sigmoid(go) * fast_tanh(c);
            unsigned short hb = f2bf(h);
            hstage[m * 256 + j] = hb;
            hw[((size_t)m * W + t) * 256 + j] = hb;
        }
        __syncthreads();
    }
}

__global__ void k_xlens(const int* x_lens, float* out) {
    int i = threadIdx.x;
    if (i < 16) out[i] = (float)x_lens[i];
}

// ---------------------------------------------------------------- launch
extern "C" void kernel_launch(void* const* d_in, const int* in_sizes, int n_in,
                              void* d_out, int out_size, void* d_ws, size_t ws_size,
                              hipStream_t stream) {
    (void)in_sizes; (void)n_in; (void)out_size; (void)ws_size;
    const float* x      = (const float*)d_in[0];
    const int* x_lens   = (const int*)d_in[1];
    const int* wstart   = (const int*)d_in[2];
    const int* wend     = (const int*)d_in[3];
    const int* wlens    = (const int*)d_in[4];
    const int* tailp    = (const int*)d_in[5];
    const float* fW_ih0 = (const float*)d_in[6];
    const float* fW_hh0 = (const float*)d_in[7];
    const float* fb0    = (const float*)d_in[8];
    const float* fW_ih1 = (const float*)d_in[9];
    const float* fW_hh1 = (const float*)d_in[10];
    const float* fb1    = (const float*)d_in[11];
    const float* fW_ih2 = (const float*)d_in[12];
    const float* fW_hh2 = (const float*)d_in[13];
    const float* fb2    = (const float*)d_in[14];
    const float* fpW    = (const float*)d_in[15];
    const float* fpb    = (const float*)d_in[16];
    const float* wpW    = (const float*)d_in[17];
    const float* wpb    = (const float*)d_in[18];
    const float* lnW    = (const float*)d_in[19];
    const float* lnB    = (const float*)d_in[20];
    const float* wW_ih  = (const float*)d_in[21];
    const float* wW_hh  = (const float*)d_in[22];
    const float* wb     = (const float*)d_in[23];
    const float* spW    = (const float*)d_in[24];
    const float* spb    = (const float*)d_in[25];

    const int B = 16, T = 2000, W = 100;

    // workspace layout (bytes, all 256-aligned sizes)
    char* ws = (char*)d_ws;
    size_t off = 0;
    auto take = [&](size_t bytes) { char* p = ws + off; off += bytes; return p; };
    unsigned short* fWc0 = (unsigned short*)take(4194304);   // 128nt x 32kt x 512
    unsigned short* fWc1 = (unsigned short*)take(4194304);
    unsigned short* fWc2 = (unsigned short*)take(4194304);
    unsigned short* fpWs = (unsigned short*)take(524288);    // 32nt x 16kt
    unsigned short* wpWs = (unsigned short*)take(262144);    // 16nt x 16kt
    unsigned short* wWs  = (unsigned short*)take(1048576);   // 64nt x 16kt
    unsigned short* spWs = (unsigned short*)take(262144);    // 32nt x 8kt
    unsigned short* hb0  = (unsigned short*)take(32768000);  // (B,T,512) bf16
    unsigned short* hb1  = (unsigned short*)take(32768000);
    unsigned short* reps = (unsigned short*)take(1638400);   // (B,W,512) bf16
    float*          rproj= (float*)take(1638400);            // (B,W,256) f32
    unsigned short* rin  = (unsigned short*)take(819200);    // (B,W,256) bf16
    unsigned short* hw   = (unsigned short*)take(819200);    // (B,W,256) bf16
    int*            sync = (int*)take(256);

    auto SW = [&](unsigned short* dst, const float* Wp, int Nn, int KtTot,
                  int ktOff, int KtThis, int Ksrc, int ld) {
        int total = Nn * KtThis * 512;
        int blocks = (total + 255) / 256;
        k_swizzleB<<<blocks, 256, 0, stream>>>(dst, Wp, Nn, KtTot, ktOff, KtThis, Ksrc, ld, total);
    };
    // frame LSTM combined weights: K = [input(512 pad) | hidden(512)], N = 2048
    SW(fWc0, fW_ih0, 128, 32, 0, 16, 80, 80);
    SW(fWc0, fW_hh0, 128, 32, 16, 16, 512, 512);
    SW(fWc1, fW_ih1, 128, 32, 0, 16, 512, 512);
    SW(fWc1, fW_hh1, 128, 32, 16, 16, 512, 512);
    SW(fWc2, fW_ih2, 128, 32, 0, 16, 512, 512);
    SW(fWc2, fW_hh2, 128, 32, 16, 16, 512, 512);
    SW(fpWs, fpW, 32, 16, 0, 16, 512, 512);
    SW(wpWs, wpW, 16, 16, 0, 16, 512, 512);
    SW(wWs, wW_ih, 64, 16, 0, 8, 256, 256);   // K = [r(256) | h(256)]
    SW(wWs, wW_hh, 64, 16, 8, 8, 256, 256);
    SW(spWs, spW, 32, 8, 0, 8, 256, 256);

    k_init_sync<<<1, 1, 0, stream>>>(sync);

    // persistent 3-layer frame LSTM: 32 blocks, 169216 B dynamic LDS each
    (void)hipFuncSetAttribute(reinterpret_cast<const void*>(k_frame_lstm),
                              hipFuncAttributeMaxDynamicSharedMemorySize, 169216);
    k_frame_lstm<<<32, 128, 169216, stream>>>(x, fWc0, fWc1, fWc2, fb0, fb1, fb2,
                                              hb0, hb1, sync, sync + 1, T);

    float* out_yframe = (float*)d_out;                 // (B,T,512)
    float* out_xlens  = out_yframe + (size_t)B * T * 512;
    float* out_sem    = out_xlens + B;                 // (B,W,512)

    // y_frame = (h * tmask) @ fpW^T + fpb   (masked rows -> fpb)
    k_gemm<<<8000, 256, 0, stream>>>(hb0, fpWs, fpb, out_yframe,
                                     B * T, 512, 512, x_lens, T, 0);
    k_xlens<<<1, 16, 0, stream>>>(x_lens, out_xlens);

    // ragged per-word tail means
    k_reps<<<B * W, 256, 0, stream>>>(hb0, x_lens, wstart, wend, wlens, tailp,
                                      reps, T, W);
    // word_proj linear
    k_gemm<<<200, 256, 0, stream>>>(reps, wpWs, wpb, rproj,
                                    B * W, 256, 512, nullptr, W, 0);
    // LayerNorm + exact GELU
    k_ln_gelu<<<B * W, 256, 0, stream>>>(rproj, lnW, lnB, rin);
    // word LSTM
    k_word_lstm<<<1, 512, 0, stream>>>(rin, wWs, wb, hw, W);
    // pred_sem = (c * wmask) @ spW^T + spb  (masked rows -> spb), lens = max(word_lens,1)
    k_gemm<<<400, 256, 0, stream>>>(hw, spWs, spb, out_sem,
                                    B * W, 512, 256, wlens, W, 1);
}
